// MultiHeadAttention_14551349199083
// MI455X (gfx1250) — matrix operand order repaired and checked by reading, not compiled, not run on hardware
//
#include <hip/hip_runtime.h>
#include <hip/hip_bf16.h>
#include <stdint.h>

// ---------- types ----------
typedef __attribute__((ext_vector_type(16))) _Float16 v16h;
typedef __attribute__((ext_vector_type(8)))  float    v8f;
typedef __attribute__((ext_vector_type(4)))  int      v4i_t;

union U16H { v16h v; _Float16 h[16]; uint32_t u[8]; uint4 q[2]; };

static __device__ __forceinline__ v8f wmma16(v16h a, v16h b, v8f c) {
  // D = A(16x32 f16) * B(32x16 f16) + C(16x16 f32)
  return __builtin_amdgcn_wmma_f32_16x16x32_f16(false, a, false, b, (short)0, c, false, false);
}

static __device__ __forceinline__ uint32_t pack2(float lo, float hi) {
  union { _Float16 h[2]; uint32_t u; } t;
  t.h[0] = (_Float16)lo; t.h[1] = (_Float16)hi;
  return t.u;
}

// ---------- CDNA5 async global->LDS copy (guarded; fallback = vgpr round trip) ----------
#if defined(__has_builtin)
#  if __has_builtin(__builtin_amdgcn_global_load_async_to_lds_b128)
#    define HAVE_ASYNC_LDS 1
#  endif
#endif

#ifdef HAVE_ASYNC_LDS
typedef __attribute__((address_space(1))) v4i_t* gptr_v4i;
typedef __attribute__((address_space(3))) v4i_t* lptr_v4i;
static __device__ __forceinline__ void async_copy16(void* lds_dst, const void* gsrc) {
  __builtin_amdgcn_global_load_async_to_lds_b128((gptr_v4i)gsrc, (lptr_v4i)lds_dst, 0, 0);
}
// wait until at most the 2 most-recent async copies are still in flight
static __device__ __forceinline__ void async_wait_le2() {
#  if __has_builtin(__builtin_amdgcn_s_wait_asynccnt)
  __builtin_amdgcn_s_wait_asynccnt(2);
#  else
  asm volatile("s_wait_asynccnt 0x2" ::: "memory");
#  endif
}
static __device__ __forceinline__ void async_wait0() {
#  if __has_builtin(__builtin_amdgcn_s_wait_asynccnt)
  __builtin_amdgcn_s_wait_asynccnt(0);
#  else
  asm volatile("s_wait_asynccnt 0x0" ::: "memory");
#  endif
}
#else
static __device__ __forceinline__ void async_copy16(void* lds_dst, const void* gsrc) {
  *(uint4*)lds_dst = *(const uint4*)gsrc;
}
static __device__ __forceinline__ void async_wait_le2() {}
static __device__ __forceinline__ void async_wait0() {}
#endif

// ---------- problem constants ----------
constexpr int D  = 64;
constexpr int H  = 16;
constexpr int BS = 4;
constexpr int QL = 2048;
constexpr int KL = 2048;

// workspace offsets (in _Float16 units)
constexpr size_t OFF_WQT  = 0;                       // 1024x64
constexpr size_t OFF_WKT  = 64 * 1024;               // 1024x64
constexpr size_t OFF_WVT  = 2 * 64 * 1024;           // 1024x64
constexpr size_t OFF_WFCT = 3 * 64 * 1024;           // 64x64
constexpr size_t OFF_Q    = OFF_WFCT + 64 * 64;      // [b,h,q,64]
constexpr size_t NQKV     = (size_t)BS * H * QL * D; // 8388608
constexpr size_t OFF_K    = OFF_Q + NQKV;            // [b,h,k,64]
constexpr size_t OFF_VT   = OFF_K + NQKV;            // [b,h,64,k]  (transposed V)
constexpr size_t OFF_O    = OFF_VT + NQKV;           // [b,q,h,64]

// ---------- kernel 0: W (KxN f32) -> W^T (NxK f16) ----------
__global__ void tcvt_kernel(const float* __restrict__ src, _Float16* __restrict__ dst,
                            int K, int N) {
  int i = blockIdx.x * blockDim.x + threadIdx.x;
  if (i >= K * N) return;
  int ncol = i / K, k = i - ncol * K;
  dst[i] = (_Float16)src[(size_t)k * N + ncol];
}

// ---------- kernel 1: projections Q = x*Wq, K = x*Wk, V = x*Wv (WMMA) ----------
__global__ __launch_bounds__(256) void proj_kernel(
    const float* __restrict__ qsrc, const float* __restrict__ ksrc,
    const _Float16* __restrict__ WqT, const _Float16* __restrict__ WkT,
    const _Float16* __restrict__ WvT,
    _Float16* __restrict__ Qh, _Float16* __restrict__ Kh, _Float16* __restrict__ Vt) {
  const int z = blockIdx.y;                      // 0=Q 1=K 2=V
  const float* src = (z == 0) ? qsrc : ksrc;
  const _Float16* WT = (z == 0) ? WqT : (z == 1) ? WkT : WvT;

  const int wave = threadIdx.x >> 5, lane = threadIdx.x & 31;
  const int g = lane >> 4, n = lane & 15;
  const int g8 = g * 8, g16 = g * 16;

  const int tid = blockIdx.x * 8 + wave;         // 0..32767 tiles
  const int mt = tid >> 6;                       // 512 row tiles over bs*ql
  const int nt = tid & 63;                       // 64 col tiles over 1024

  // A operand: rows m = lane&15, two 8-half chunks per kstep (fp32 -> f16)
  const float* ar = src + (size_t)(mt * 16 + n) * D;
  U16H a[2];
  #pragma unroll
  for (int ks = 0; ks < 2; ++ks) {
    #pragma unroll
    for (int j = 0; j < 2; ++j) {
      const int base = ks * 32 + g16 + j * 8;
      float4 f0 = *(const float4*)(ar + base);
      float4 f1 = *(const float4*)(ar + base + 4);
      a[ks].h[j*8+0] = (_Float16)f0.x; a[ks].h[j*8+1] = (_Float16)f0.y;
      a[ks].h[j*8+2] = (_Float16)f0.z; a[ks].h[j*8+3] = (_Float16)f0.w;
      a[ks].h[j*8+4] = (_Float16)f1.x; a[ks].h[j*8+5] = (_Float16)f1.y;
      a[ks].h[j*8+6] = (_Float16)f1.z; a[ks].h[j*8+7] = (_Float16)f1.w;
    }
  }

  const int col = nt * 16 + n;                   // output column 0..1023
  v8f acc = {};
  #pragma unroll
  for (int ks = 0; ks < 2; ++ks) {
    v16h bv = *(const v16h*)(WT + (size_t)col * D + ks * 32 + g16);
    acc = wmma16(a[ks].v, bv, acc);
  }

  const int h  = col >> 6;                       // head
  const int dd = col & 63;                       // dim within head
  const int rowD0 = mt * 16 + g8;                // first of 8 result rows
  const int b   = rowD0 >> 11;
  const int qr0 = rowD0 & (QL - 1);

  if (z == 2) {
    // V stored transposed: Vt[b,h,dd, qr0..qr0+7] -> one b128 store
    union { uint32_t u[4]; uint4 q; } pk;
    #pragma unroll
    for (int i = 0; i < 4; ++i) pk.u[i] = pack2(acc[2*i], acc[2*i+1]);
    *(uint4*)(Vt + (((size_t)(b * H + h)) * D + dd) * KL + qr0) = pk.q;
  } else {
    _Float16* dst = (z == 0) ? Qh : Kh;
    _Float16* dp = dst + (((size_t)(b * H + h)) * QL + qr0) * D + dd;
    #pragma unroll
    for (int r = 0; r < 8; ++r) dp[(size_t)r * D] = (_Float16)acc[r];
  }
}

// ---------- kernel 2: flash attention (WMMA + double-buffered async LDS staging) ----------
__global__ __launch_bounds__(256) void attn_kernel(
    const _Float16* __restrict__ Qh, const _Float16* __restrict__ Kh,
    const _Float16* __restrict__ Vt, const unsigned char* __restrict__ mask,
    _Float16* __restrict__ Oh) {
  __shared__ _Float16 Ks[2][32][72];   // 2 x (32 keys x 64 dims, +pad)
  __shared__ _Float16 Vs[2][64][40];   // 2 x (64 dims x 32 keys, +pad), from Vt

  const int wave = threadIdx.x >> 5, lane = threadIdx.x & 31;
  const int g = lane >> 4, qn = lane & 15;
  const int g8 = g * 8, g16 = g * 16;

  const int bh = blockIdx.x >> 4;                // b*H + h
  const int b  = bh >> 4;
  const int h  = bh & 15;
  const int q  = (blockIdx.x & 15) * 128 + wave * 16 + qn;  // this lane's query

  // Q^T B-operands (loop invariant): halves = d index g*16+h
  const _Float16* qrow = Qh + ((size_t)bh * QL + q) * D;
  U16H bq[2];
  bq[0].v = *(const v16h*)(qrow + g16);
  bq[1].v = *(const v16h*)(qrow + 32 + g16);

  v8f o[4] = {};                                 // O^T accum: 4 d-tiles x 8 rows
  float Mrun = -1e30f, Lrun = 0.0f;
  const unsigned char* mrow = mask + ((size_t)b * QL + q) * KL;

  // per-thread staging addresses (constant across the key loop)
  const int kr = threadIdx.x >> 3, kc = (threadIdx.x & 7) * 8;
  const int vr = threadIdx.x >> 2, vc = (threadIdx.x & 3) * 8;
  const _Float16* kgp = Kh + ((size_t)bh * KL + kr) * D + kc;
  const _Float16* vgp = Vt + ((size_t)bh * D + vr) * KL + vc;

  // prologue: start DMA for chunk 0 into buffer 0
  async_copy16(&Ks[0][kr][kc], kgp);
  async_copy16(&Vs[0][vr][vc], vgp);

  for (int c = 0; c < KL; c += 32) {
    const int buf = (c >> 5) & 1;
    if (c + 32 < KL) {
      // kick off next chunk's DMA into the other buffer, then wait for current
      async_copy16(&Ks[buf ^ 1][kr][kc], kgp + (size_t)(c + 32) * D);
      async_copy16(&Vs[buf ^ 1][vr][vc], vgp + (c + 32));
      __builtin_prefetch(mrow + c + 32, 0, 1);
      async_wait_le2();
    } else {
      async_wait0();
    }
    __syncthreads();                             // current buffer visible to all waves

    // S^T = K * Q^T : two 16-key tiles
    float s[2][8];
    #pragma unroll
    for (int t = 0; t < 2; ++t) {
      v8f st = {};
      #pragma unroll
      for (int ks = 0; ks < 2; ++ks) {
        U16H a;
        a.q[0] = *(const uint4*)&Ks[buf][t * 16 + qn][ks * 32 + g16];
        a.q[1] = *(const uint4*)&Ks[buf][t * 16 + qn][ks * 32 + g16 + 8];
        st = wmma16(a.v, bq[ks].v, st);
      }
      // mask + scale; lane's 8 keys are contiguous bytes
      unsigned long long mv =
          *(const unsigned long long*)(mrow + c + t * 16 + g8);
      #pragma unroll
      for (int r = 0; r < 8; ++r) {
        float x = st[r] * 0.125f;                // 1/sqrt(64)
        if ((mv >> (8 * r)) & 0xffull) x = -1e30f;
        s[t][r] = x;
      }
    }

    // online softmax over this chunk's 32 keys (16 in-lane + pair lane)
    float mloc = -1e30f;
    #pragma unroll
    for (int t = 0; t < 2; ++t)
      #pragma unroll
      for (int r = 0; r < 8; ++r) mloc = fmaxf(mloc, s[t][r]);
    mloc = fmaxf(mloc, __shfl_xor(mloc, 16));
    const float Mn = fmaxf(Mrun, mloc);
    const float alpha = __expf(Mrun - Mn);
    Mrun = Mn;

    float ls = 0.0f;
    uint32_t pk0[4], pk1[4];
    #pragma unroll
    for (int i = 0; i < 4; ++i) {
      float a0 = __expf(s[0][2*i] - Mn), a1 = __expf(s[0][2*i+1] - Mn);
      float c0 = __expf(s[1][2*i] - Mn), c1 = __expf(s[1][2*i+1] - Mn);
      ls += a0 + a1 + c0 + c1;
      pk0[i] = pack2(a0, a1);
      pk1[i] = pack2(c0, c1);
    }
    ls += __shfl_xor(ls, 16);
    Lrun = Lrun * alpha + ls;
    #pragma unroll
    for (int dt = 0; dt < 4; ++dt)
      #pragma unroll
      for (int r = 0; r < 8; ++r) o[dt][r] *= alpha;

    // assemble P^T B-operand via pair-lane exchanges (C-layout -> B-layout)
    U16H bp;
    #pragma unroll
    for (int i = 0; i < 4; ++i) {
      uint32_t o0 = __shfl_xor(pk0[i], 16);
      uint32_t o1 = __shfl_xor(pk1[i], 16);
      bp.u[i]     = g ? o1 : pk0[i];
      bp.u[4 + i] = g ? pk1[i] : o0;
    }

    // O^T += V^T * P^T : 4 d-tiles
    #pragma unroll
    for (int dt = 0; dt < 4; ++dt) {
      U16H a;
      a.q[0] = *(const uint4*)&Vs[buf][dt * 16 + qn][g16];
      a.q[1] = *(const uint4*)&Vs[buf][dt * 16 + qn][g16 + 8];
      o[dt] = wmma16(a.v, bp.v, o[dt]);
    }

    __syncthreads();  // all waves done reading buf before its DMA overwrite next iter
  }

  // finalize: divide by L; fully-masked rows -> 0 (nan_to_num)
  float inv = (Lrun > 0.0f) ? (1.0f / Lrun) : 0.0f;
  if (Mrun <= -0.9e30f) inv = 0.0f;
  #pragma unroll
  for (int dt = 0; dt < 4; ++dt) {
    union { uint32_t u[4]; uint4 qv; } pk;
    #pragma unroll
    for (int i = 0; i < 4; ++i)
      pk.u[i] = pack2(o[dt][2*i] * inv, o[dt][2*i+1] * inv);
    *(uint4*)(Oh + (((size_t)b * QL + q) * H + h) * D + dt * 16 + g8) = pk.qv;
  }
}

// ---------- kernel 3: out = tanh(O @ Wfc + bfc) ----------
__global__ __launch_bounds__(256) void fc_kernel(
    const _Float16* __restrict__ Oh, const _Float16* __restrict__ WfcT,
    const float* __restrict__ bfc, float* __restrict__ out) {
  const int wave = threadIdx.x >> 5, lane = threadIdx.x & 31;
  const int g = lane >> 4, n = lane & 15;
  const int g8 = g * 8, g16 = g * 16;

  const int tile = blockIdx.x * 8 + wave;        // 8192 row tiles
  const size_t row0 = (size_t)tile * 16;

  const _Float16* ar = Oh + (row0 + n) * D;
  U16H a[2];
  #pragma unroll
  for (int ks = 0; ks < 2; ++ks) {
    a[ks].q[0] = *(const uint4*)(ar + ks * 32 + g16);
    a[ks].q[1] = *(const uint4*)(ar + ks * 32 + g16 + 8);
  }

  #pragma unroll
  for (int nt = 0; nt < 4; ++nt) {
    const int col = nt * 16 + n;
    v8f acc = {};
    #pragma unroll
    for (int ks = 0; ks < 2; ++ks) {
      v16h bv = *(const v16h*)(WfcT + (size_t)col * D + ks * 32 + g16);
      acc = wmma16(a[ks].v, bv, acc);
    }
    const float bb = bfc[col];
    #pragma unroll
    for (int r = 0; r < 8; ++r)
      out[(row0 + g8 + r) * D + col] = tanhf(acc[r] + bb);
  }
}

// ---------- launch ----------
extern "C" void kernel_launch(void* const* d_in, const int* in_sizes, int n_in,
                              void* d_out, int out_size, void* d_ws, size_t ws_size,
                              hipStream_t stream) {
  (void)in_sizes; (void)n_in; (void)out_size; (void)ws_size;
  const float* q_origin = (const float*)d_in[0];
  const float* k_origin = (const float*)d_in[1];
  const unsigned char* mask = (const unsigned char*)d_in[2];
  const float* Wq  = (const float*)d_in[3];
  const float* Wk  = (const float*)d_in[4];
  const float* Wv  = (const float*)d_in[5];
  const float* Wfc = (const float*)d_in[6];
  const float* bfc = (const float*)d_in[7];

  _Float16* ws   = (_Float16*)d_ws;
  _Float16* WqT  = ws + OFF_WQT;
  _Float16* WkT  = ws + OFF_WKT;
  _Float16* WvT  = ws + OFF_WVT;
  _Float16* WfcT = ws + OFF_WFCT;
  _Float16* Qh   = ws + OFF_Q;
  _Float16* Kh   = ws + OFF_K;
  _Float16* Vt   = ws + OFF_VT;
  _Float16* Oh   = ws + OFF_O;

  // weight transposes -> f16
  tcvt_kernel<<<(64 * 1024 + 255) / 256, 256, 0, stream>>>(Wq, WqT, 64, 1024);
  tcvt_kernel<<<(64 * 1024 + 255) / 256, 256, 0, stream>>>(Wk, WkT, 64, 1024);
  tcvt_kernel<<<(64 * 1024 + 255) / 256, 256, 0, stream>>>(Wv, WvT, 64, 1024);
  tcvt_kernel<<<(64 * 64 + 255) / 256, 256, 0, stream>>>(Wfc, WfcT, 64, 64);

  // projections: 512 m-tiles * 64 n-tiles per matrix, 8 waves/block
  proj_kernel<<<dim3(4096, 3), 256, 0, stream>>>(q_origin, k_origin,
                                                 WqT, WkT, WvT, Qh, Kh, Vt);

  // flash attention: one block per (b,h,128 queries)
  attn_kernel<<<dim3(BS * H * (QL / 128)), 256, 0, stream>>>(Qh, Kh, Vt, mask, Oh);

  // final FC + tanh: 8192 row tiles, 8 waves/block
  fc_kernel<<<dim3(1024), 256, 0, stream>>>(Oh, WfcT, bfc, (float*)d_out);
}
